// GLALayer_62302795596506
// MI455X (gfx1250) — compile-verified, matrix operand order checked by reading, not run on hardware
//
#include <hip/hip_runtime.h>
#include <hip/hip_bf16.h>

#define BB 4
#define TT 4096
#define DM 1024
#define NH 8
#define DH 64
#define DI 512
#define BT (BB*TT)

typedef __attribute__((ext_vector_type(16))) _Float16 v16h;
typedef __attribute__((ext_vector_type(8)))  _Float16 h8;
typedef __attribute__((ext_vector_type(8)))  float    v8f;
typedef int v4i_vs __attribute__((vector_size(16)));   // matches builtin's pointee type

// ---- CDNA5 async global->LDS path (guarded; falls back to sync staging) ----
#if defined(__has_builtin)
#if __has_builtin(__builtin_amdgcn_global_load_async_to_lds_b128)
#define HAVE_ASYNC 1
#warning CDNA5 probe: global_load_async_to_lds_b128 builtin FOUND - async staging enabled
#endif
#endif
#ifndef HAVE_ASYNC
#define HAVE_ASYNC 0
#warning CDNA5 probe: global_load_async_to_lds_b128 builtin NOT found - sync staging fallback
#endif

// Builtin prototype (from hipcc diagnostic): param0 = addrspace(1) int4*,
// param1 = LDS int4*, then two int immediates (offset, cpol).
#define AS1V4(p) ((__attribute__((address_space(1))) v4i_vs*)(uintptr_t)(p))
#define AS3V4(p) ((__attribute__((address_space(3))) v4i_vs*)(unsigned)(uintptr_t)(p))

#if HAVE_ASYNC
#define ASYNC_WAIT() asm volatile("s_wait_asynccnt 0x0" ::: "memory")
#else
#define ASYNC_WAIT() ((void)0)
#endif

// Copy one 8192B tile (4096 halves) global->LDS with 256 threads (32B each).
__device__ __forceinline__ void stage_tile(const _Float16* g, _Float16* l, int tid) {
#if HAVE_ASYNC
  const char* gp = (const char*)g + tid * 32;
  char* lp = (char*)l + tid * 32;
  __builtin_amdgcn_global_load_async_to_lds_b128(AS1V4(gp), AS3V4(lp), 0, 0);
  __builtin_amdgcn_global_load_async_to_lds_b128(AS1V4(gp + 16), AS3V4(lp + 16), 0, 0);
#else
  const h8* s = (const h8*)g + tid * 2;
  h8* d = (h8*)l + tid * 2;
  d[0] = s[0];
  d[1] = s[1];
#endif
}

// Load a 16-bit WMMA A/B fragment from an LDS row (row-major, leading dim LD halves).
__device__ __forceinline__ v16h frag_load(const _Float16* rowp, int kk, int ks) {
  const h8 lo = *(const h8*)(rowp + kk + 8 * ks);
  const h8 hi = *(const h8*)(rowp + kk + 16 + 8 * ks);
  v16h f;
#pragma unroll
  for (int i = 0; i < 8; ++i) { f[i] = lo[i]; f[i + 8] = hi[i]; }
  return f;
}

// ---------------- LayerNorm: x fp32 -> xn f16 -------------------------------
__global__ __launch_bounds__(256)
void ln_kernel(const float* __restrict__ x, const float* __restrict__ gamma,
               const float* __restrict__ beta, _Float16* __restrict__ xn) {
  const int row = blockIdx.x, tid = threadIdx.x;
  __shared__ float red[18];
  float vv[4], s = 0.f, s2 = 0.f;
#pragma unroll
  for (int k = 0; k < 4; ++k) {
    float v = x[(size_t)row * DM + tid + 256 * k];
    vv[k] = v; s += v; s2 += v * v;
  }
#pragma unroll
  for (int o = 16; o > 0; o >>= 1) { s += __shfl_xor(s, o, 32); s2 += __shfl_xor(s2, o, 32); }
  const int w = tid >> 5;
  if ((tid & 31) == 0) { red[w] = s; red[8 + w] = s2; }
  __syncthreads();
  if (tid == 0) {
    float ts = 0.f, t2 = 0.f;
    for (int i = 0; i < 8; ++i) { ts += red[i]; t2 += red[8 + i]; }
    float mu = ts / DM, var = t2 / DM - mu * mu;
    red[16] = mu; red[17] = rsqrtf(var + 1e-5f);
  }
  __syncthreads();
  const float mu = red[16], rstd = red[17];
#pragma unroll
  for (int k = 0; k < 4; ++k) {
    int i = tid + 256 * k;
    xn[(size_t)row * DM + i] = (_Float16)((vv[k] - mu) * rstd * gamma[i] + beta[i]);
  }
}

// ---------------- Weight cast fp32 -> f16 (Wq|Wk|Wv|Wa packed, Wo) ----------
__global__ void wconv_kernel(const float* __restrict__ Wq, const float* __restrict__ Wk,
                             const float* __restrict__ Wv, const float* __restrict__ Wa,
                             const float* __restrict__ Wo, _Float16* __restrict__ Wcat,
                             _Float16* __restrict__ Wo16) {
  int idx = blockIdx.x * 256 + threadIdx.x;
  const int N1 = DM * 4 * DI;
  if (idx < N1) {
    int k = idx >> 11, c = idx & 2047;
    int sel = c >> 9, cc = c & 511;
    const float* W = (sel == 0) ? Wq : (sel == 1) ? Wk : (sel == 2) ? Wv : Wa;
    Wcat[idx] = (_Float16)W[k * DI + cc];
  } else {
    int j = idx - N1;
    if (j < DI * DM) Wo16[j] = (_Float16)Wo[j];
  }
}

// ---------------- WMMA GEMM: C(MxN) = A(MxK f16) @ B(KxN f16) [+resid] ------
template <bool RES, typename OutT>
__global__ __launch_bounds__(256)
void gemm_f16(const _Float16* __restrict__ A, const _Float16* __restrict__ Bm,
              OutT* __restrict__ C, const float* __restrict__ resid,
              int M, int N, int K) {
  constexpr int LDA = 40;                       // 32 + 8 halves pad, keeps 16B align
  __shared__ alignas(16) _Float16 As[128 * LDA];
  __shared__ alignas(16) _Float16 Bt[128 * LDA]; // B tile transposed: [n][k]
  const int tid = threadIdx.x;
  const int lane = tid & 31, w = tid >> 5;
  const int lm = lane & 15, ks = lane >> 4;
  const int m0 = blockIdx.y * 128, n0 = blockIdx.x * 128;
  const int wr = w & 3, wc = w >> 2;            // 4x2 wave grid -> 32x64 per wave
  v8f acc[2][4];
#pragma unroll
  for (int a = 0; a < 2; ++a)
#pragma unroll
    for (int b = 0; b < 4; ++b)
#pragma unroll
      for (int e = 0; e < 8; ++e) acc[a][b][e] = 0.f;

  const int arow = tid >> 1, ahs = tid & 1;     // A loader: 2 threads/row, 16 halves each
  const int bkr = tid >> 3, bcg = tid & 7;      // B loader: k-row, 16-col group

  for (int kk = 0; kk < K; kk += 32) {
    const _Float16* ga = A + (size_t)(m0 + arow) * K + kk + ahs * 16;
    _Float16* la = &As[arow * LDA + ahs * 16];
#if HAVE_ASYNC
    __builtin_amdgcn_global_load_async_to_lds_b128(AS1V4(ga), AS3V4(la), 0, 0);
    __builtin_amdgcn_global_load_async_to_lds_b128(AS1V4((const char*)ga + 16),
                                                   AS3V4((char*)la + 16), 0, 0);
#else
    h8 a0 = *(const h8*)(ga);
    h8 a1 = *(const h8*)(ga + 8);
    *(h8*)(la) = a0;
    *(h8*)(la + 8) = a1;
#endif

    const _Float16* gb = Bm + (size_t)(kk + bkr) * N + n0 + bcg * 16;
    h8 b0 = *(const h8*)(gb);
    h8 b1 = *(const h8*)(gb + 8);
#pragma unroll
    for (int i = 0; i < 8; ++i) {
      Bt[(bcg * 16 + i) * LDA + bkr]     = b0[i];
      Bt[(bcg * 16 + 8 + i) * LDA + bkr] = b1[i];
    }
    if (kk + 32 < K) {                           // global_prefetch_b8 hints
      __builtin_prefetch(ga + 32, 0, 0);
      __builtin_prefetch(Bm + (size_t)(kk + 32 + bkr) * N + n0 + bcg * 16, 0, 0);
    }
    ASYNC_WAIT();
    __syncthreads();

    v16h af[2], bf[4];
#pragma unroll
    for (int mt = 0; mt < 2; ++mt)
      af[mt] = frag_load(&As[(wr * 32 + mt * 16 + lm) * LDA], 0, ks);
#pragma unroll
    for (int nt = 0; nt < 4; ++nt)
      bf[nt] = frag_load(&Bt[(wc * 64 + nt * 16 + lm) * LDA], 0, ks);
#pragma unroll
    for (int mt = 0; mt < 2; ++mt)
#pragma unroll
      for (int nt = 0; nt < 4; ++nt)
        acc[mt][nt] = __builtin_amdgcn_wmma_f32_16x16x32_f16(
            false, af[mt], false, bf[nt], (short)0, acc[mt][nt], false, false);
    __syncthreads();
  }

#pragma unroll
  for (int mt = 0; mt < 2; ++mt)
#pragma unroll
    for (int nt = 0; nt < 4; ++nt) {
      int n = n0 + wc * 64 + nt * 16 + lm;
#pragma unroll
      for (int r = 0; r < 8; ++r) {
        int m = m0 + wr * 32 + mt * 16 + r + 8 * ks;
        float v = acc[mt][nt][r];
        if (RES) v += resid[(size_t)m * N + n];
        C[(size_t)m * N + n] = (OutT)v;
      }
    }
}

// ---------------- Epilogue: split proj, normalize k, sigmoid gate -> log ----
__global__ __launch_bounds__(256)
void post_kernel(const _Float16* __restrict__ proj, const float* __restrict__ ba,
                 _Float16* __restrict__ q16, _Float16* __restrict__ k16,
                 _Float16* __restrict__ v16, _Float16* __restrict__ loga) {
  const int row = blockIdx.x * 8 + (threadIdx.x >> 5);   // (bt, h)
  const int lane = threadIdx.x & 31;
  const int bt = row >> 3, h = row & 7;
  const int b = bt / TT, t = bt % TT;
  const _Float16* pr = proj + (size_t)bt * 2048 + h * 64;
  float k0 = (float)pr[512 + 2 * lane], k1 = (float)pr[512 + 2 * lane + 1];
  float ss = k0 * k0 + k1 * k1;
#pragma unroll
  for (int o = 16; o > 0; o >>= 1) ss += __shfl_xor(ss, o, 32);
  const float inv = 1.f / fmaxf(sqrtf(ss), 1e-12f);
  const size_t dst = ((size_t)(b * NH + h) * TT + t) * 64;
#pragma unroll
  for (int u = 0; u < 2; ++u) {
    int d = 2 * lane + u;
    q16[dst + d] = pr[d];
    k16[dst + d] = (_Float16)((u ? k1 : k0) * inv);
    v16[dst + d] = pr[1024 + d];
    float av = (float)pr[1536 + d] + ba[h * 64 + d];
    float sg = 1.f / (1.f + __expf(-av));
    loga[dst + d] = (_Float16)__logf(sg);
  }
}

// ---------------- Chunked GLA recurrence (chunk C=64, all WMMA) -------------
// Dynamic LDS layout (halves):
//   sQ 0, sKb 4608, sVT 9216, sP 13824, sST 18432, sS 23040,
//   stg 27648 (= [2 bufs][4 tiles: q,k,v,loga][4096]), sTot(f32[64]) @60416
#define GLA_SMEM_BYTES ((60416 + 128) * 2)

__global__ __launch_bounds__(256)
void gla_kernel(const _Float16* __restrict__ q16, const _Float16* __restrict__ k16,
                const _Float16* __restrict__ v16, const _Float16* __restrict__ loga,
                _Float16* __restrict__ o16) {
  extern __shared__ _Float16 smem[];
  _Float16* sQ  = smem;            // Qb[t][j]  = q * e^{cum}
  _Float16* sKb = smem + 4608;     // Kb[t][j]  = k * e^{-cum}
  _Float16* sVT = smem + 9216;     // V^T[n][t]
  _Float16* sP  = smem + 13824;    // tril(Qb Kb^T)[t][s]
  _Float16* sST = smem + 18432;    // S^T[n][j]
  _Float16* sS  = smem + 23040;    // S[j][n]
  _Float16* stg = smem + 27648;    // staging: [buf][4][4096]
  float*    sTot = (float*)(smem + 60416);

  const int tid = threadIdx.x, lane = tid & 31, w = tid >> 5;
  const int lm = lane & 15, ks = lane >> 4;
  const int bh = blockIdx.x, b = bh >> 3, h = bh & 7;
  const int mt = w & 3;                          // 16-row tile index
  const int ntb = (w >> 2) * 2;                  // two 16-col tiles per wave
  for (int i = tid; i < 4608; i += 256) { sS[i] = (_Float16)0.f; sST[i] = (_Float16)0.f; }
  const size_t base0 = (size_t)bh * TT * 64;

  // stage chunk 0 into buffer 0
  {
    _Float16* d = stg;
    stage_tile(q16 + base0, d, tid);
    stage_tile(k16 + base0, d + 4096, tid);
    stage_tile(v16 + base0, d + 2 * 4096, tid);
    stage_tile(loga + base0, d + 3 * 4096, tid);
  }

  for (int ch = 0; ch < TT / 64; ++ch) {
    const int buf = ch & 1;
    _Float16* st = stg + buf * 4 * 4096;
    ASYNC_WAIT();
    __syncthreads();                 // staged chunk visible; prior phases done
    if (ch + 1 < TT / 64) {          // prefetch next chunk into other buffer
      const size_t nb = base0 + (size_t)(ch + 1) * 4096;
      _Float16* d = stg + (buf ^ 1) * 4 * 4096;
      stage_tile(q16 + nb, d, tid);
      stage_tile(k16 + nb, d + 4096, tid);
      stage_tile(v16 + nb, d + 2 * 4096, tid);
      stage_tile(loga + nb, d + 3 * 4096, tid);
    }
    _Float16* stQ = st;
    _Float16* stK = st + 4096;
    _Float16* stV = st + 2 * 4096;
    _Float16* stL = st + 3 * 4096;
    // A: per-dim prefix sums of log-gates, in place on the staged tile
    if (tid < 64) {
      float c = 0.f;
      for (int t = 0; t < 64; ++t) {
        c += (float)stL[t * 64 + tid];
        stL[t * 64 + tid] = (_Float16)c;
      }
      sTot[tid] = c;
    }
    __syncthreads();
    // B: build chunk tiles
#pragma unroll
    for (int i = 0; i < 16; ++i) {
      int e = i * 256 + tid, t = e >> 6, j = e & 63;
      float cu = (float)stL[e];
      float qv = (float)stQ[e];
      float kv = (float)stK[e];
      sQ [t * 72 + j] = (_Float16)(qv * __expf(cu));
      sKb[t * 72 + j] = (_Float16)(kv * __expf(-cu));
      sVT[j * 72 + t] = stV[e];
    }
    __syncthreads();
    // C: P = tril(Qb @ Kb^T)
#pragma unroll
    for (int tt = 0; tt < 2; ++tt) {
      v8f acc;
#pragma unroll
      for (int e = 0; e < 8; ++e) acc[e] = 0.f;
#pragma unroll
      for (int kk = 0; kk < 64; kk += 32) {
        v16h a  = frag_load(&sQ [(mt * 16 + lm) * 72], kk, ks);
        v16h bb = frag_load(&sKb[((ntb + tt) * 16 + lm) * 72], kk, ks);
        acc = __builtin_amdgcn_wmma_f32_16x16x32_f16(false, a, false, bb, (short)0, acc, false, false);
      }
#pragma unroll
      for (int r = 0; r < 8; ++r) {
        int tl = mt * 16 + r + 8 * ks, sl = (ntb + tt) * 16 + lm;
        sP[tl * 72 + sl] = (_Float16)((sl <= tl) ? acc[r] : 0.f);
      }
    }
    __syncthreads();
    // D: O = P @ V + Qb @ S
#pragma unroll
    for (int tt = 0; tt < 2; ++tt) {
      v8f acc;
#pragma unroll
      for (int e = 0; e < 8; ++e) acc[e] = 0.f;
#pragma unroll
      for (int kk = 0; kk < 64; kk += 32) {
        v16h a  = frag_load(&sP [(mt * 16 + lm) * 72], kk, ks);
        v16h bb = frag_load(&sVT[((ntb + tt) * 16 + lm) * 72], kk, ks);
        acc = __builtin_amdgcn_wmma_f32_16x16x32_f16(false, a, false, bb, (short)0, acc, false, false);
      }
#pragma unroll
      for (int kk = 0; kk < 64; kk += 32) {
        v16h a  = frag_load(&sQ [(mt * 16 + lm) * 72], kk, ks);
        v16h bb = frag_load(&sST[((ntb + tt) * 16 + lm) * 72], kk, ks);
        acc = __builtin_amdgcn_wmma_f32_16x16x32_f16(false, a, false, bb, (short)0, acc, false, false);
      }
      const size_t ob = ((size_t)b * TT + ch * 64) * DI + h * 64;
#pragma unroll
      for (int r = 0; r < 8; ++r) {
        int tl = mt * 16 + r + 8 * ks, nl = (ntb + tt) * 16 + lm;
        o16[ob + (size_t)tl * DI + nl] = (_Float16)acc[r];
      }
    }
    __syncthreads();
    // E: S = diag(e^{tot}) S + Kd^T @ V,  Kd[t][j] = Kb[t][j] * e^{tot[j]}
#pragma unroll
    for (int tt = 0; tt < 2; ++tt) {
      v8f acc;
#pragma unroll
      for (int e = 0; e < 8; ++e) acc[e] = 0.f;
      const int jrow = mt * 16 + lm;
      const float cs = __expf(sTot[jrow]);
#pragma unroll
      for (int kk = 0; kk < 64; kk += 32) {
        v16h a;
#pragma unroll
        for (int i = 0; i < 8; ++i) {
          int t0 = kk + 8 * ks + i;
          int t1 = kk + 16 + 8 * ks + i;
          a[i]     = (_Float16)((float)sKb[t0 * 72 + jrow] * cs);
          a[i + 8] = (_Float16)((float)sKb[t1 * 72 + jrow] * cs);
        }
        v16h bb = frag_load(&sVT[((ntb + tt) * 16 + lm) * 72], kk, ks);
        acc = __builtin_amdgcn_wmma_f32_16x16x32_f16(false, a, false, bb, (short)0, acc, false, false);
      }
#pragma unroll
      for (int r = 0; r < 8; ++r) {
        int jl = mt * 16 + r + 8 * ks, nl = (ntb + tt) * 16 + lm;
        float ev = __expf(sTot[jl]);
        float sv = ev * (float)sS[jl * 72 + nl] + acc[r];
        sS[jl * 72 + nl]  = (_Float16)sv;
        sST[nl * 72 + jl] = (_Float16)sv;
      }
    }
    __syncthreads();
  }
}

// ---------------------------------------------------------------------------
extern "C" void kernel_launch(void* const* d_in, const int* in_sizes, int n_in,
                              void* d_out, int out_size, void* d_ws, size_t ws_size,
                              hipStream_t stream) {
  (void)in_sizes; (void)n_in; (void)out_size; (void)ws_size;
  const float* x     = (const float*)d_in[0];
  const float* Wq    = (const float*)d_in[1];
  const float* Wk    = (const float*)d_in[2];
  const float* Wv    = (const float*)d_in[3];
  const float* Wa    = (const float*)d_in[4];
  const float* ba    = (const float*)d_in[5];
  const float* Wo    = (const float*)d_in[6];
  const float* gamma = (const float*)d_in[7];
  const float* beta  = (const float*)d_in[8];
  float* out = (float*)d_out;

  char* ws = (char*)d_ws;
  size_t off = 0;
  auto alloc = [&](size_t bytes) -> char* {
    char* p = ws + off;
    off = (off + bytes + 255) & ~(size_t)255;
    return p;
  };
  _Float16* xn16   = (_Float16*)alloc((size_t)BT * DM * 2);
  _Float16* Wcat16 = (_Float16*)alloc((size_t)DM * 4 * DI * 2);
  _Float16* Wo16   = (_Float16*)alloc((size_t)DI * DM * 2);
  _Float16* proj16 = (_Float16*)alloc((size_t)BT * 4 * DI * 2);
  _Float16* q16    = (_Float16*)alloc((size_t)BT * DI * 2);
  _Float16* k16    = (_Float16*)alloc((size_t)BT * DI * 2);
  _Float16* v16    = (_Float16*)alloc((size_t)BT * DI * 2);
  _Float16* loga16 = (_Float16*)alloc((size_t)BT * DI * 2);
  _Float16* o16    = (_Float16*)alloc((size_t)BT * DI * 2);

  ln_kernel<<<BT, 256, 0, stream>>>(x, gamma, beta, xn16);
  const int nw = DM * 4 * DI + DI * DM;
  wconv_kernel<<<(nw + 255) / 256, 256, 0, stream>>>(Wq, Wk, Wv, Wa, Wo, Wcat16, Wo16);
  gemm_f16<false, _Float16><<<dim3((4 * DI) / 128, BT / 128), 256, 0, stream>>>(
      xn16, Wcat16, proj16, nullptr, BT, 4 * DI, DM);
  post_kernel<<<BT * NH / 8, 256, 0, stream>>>(proj16, ba, q16, k16, v16, loga16);
  gla_kernel<<<BB * NH, 256, GLA_SMEM_BYTES, stream>>>(q16, k16, v16, loga16, o16);
  gemm_f16<true, float><<<dim3(DM / 128, BT / 128), 256, 0, stream>>>(
      o16, Wo16, out, x, BT, DM, DI);
}
